// my_local_attention_49331994362591
// MI455X (gfx1250) — matrix operand order repaired and checked by reading, not compiled, not run on hardware
//
#include <hip/hip_runtime.h>

typedef __attribute__((ext_vector_type(16))) _Float16 v16h;
typedef __attribute__((ext_vector_type(8)))  float    v8f;

#define HH 512
#define WW 512
#define CC 64
#define MID 32
#define NOUT 128
#define TILE 14
#define HALO 16
#define HWSZ (HH*WW)

// LDS strides in halves (padded to spread banks)
#define XT_STR 72
#define WT_STR 72
#define X1_STR 36
#define X3_STR 72

// LDS byte offsets
#define OFF_XT   0                      // 256*72*2   = 36864
#define OFF_WT   36864                  // 128*72*2   = 18432
#define OFF_X1   55296                  // 256*36*2   = 18432
#define OFF_X2   73728                  // 256*36*2   = 18432
#define OFF_X3   92160                  // 256*72*2   = 36864
#define OFF_BIAS 129024                 // 128*4      = 512
#define SMEM_BYTES 129536

union AB16 { uint4 u[2]; v16h v; };

extern "C" __global__ __launch_bounds__(256)
void fused_local_attn_gfx1250(const float* __restrict__ x,
                              const float* __restrict__ w1, const float* __restrict__ b1,
                              const float* __restrict__ w2, const float* __restrict__ b2,
                              const float* __restrict__ w3, const float* __restrict__ b3,
                              const float* __restrict__ gamma,
                              float* __restrict__ out)
{
    extern __shared__ char smem[];
    _Float16* XtL  = (_Float16*)(smem + OFF_XT);   // [256 pixels][72] halo tile, K contiguous
    _Float16* WtL  = (_Float16*)(smem + OFF_WT);   // [128 outch][72]   weights, K contiguous
    _Float16* X1L  = (_Float16*)(smem + OFF_X1);   // [256][36]
    _Float16* X2L  = (_Float16*)(smem + OFF_X2);   // [256][36]
    _Float16* X3L  = (_Float16*)(smem + OFF_X3);   // [256][72]
    float*    biasL= (float*)  (smem + OFF_BIAS);  // [128]

    const int tid = threadIdx.x;
    const int tx = blockIdx.x, ty = blockIdx.y, bb = blockIdx.z;

    // ---------- phase 0: stage bias, weights, x halo tile (fp32 -> f16) ----------
    if (tid < NOUT) {
        float v = (tid < MID) ? b1[tid] : (tid < 2*MID) ? b2[tid - MID] : b3[tid - 2*MID];
        biasL[tid] = v;
    }
    for (int idx = tid; idx < NOUT*CC; idx += 256) {
        int n = idx >> 6, k = idx & 63;
        float wv = (n < MID)   ? w1[n*CC + k]
                 : (n < 2*MID) ? w2[(n - MID)*CC + k]
                 :               w3[(n - 2*MID)*CC + k];
        WtL[n*WT_STR + k] = (_Float16)wv;
    }
    {
        const int p  = tid;                       // 256 threads <-> 256 halo pixels
        const int py = p >> 4, px = p & 15;
        const int gh = ty*TILE - 1 + py, gw = tx*TILE - 1 + px;
        const bool ok = (gh >= 0) & (gh < HH) & (gw >= 0) & (gw < WW);
        const int ch = ok ? gh : 0, cw = ok ? gw : 0;
        const float* xp = x + (size_t)bb*CC*HWSZ + (size_t)ch*WW + cw;
        #pragma unroll 4
        for (int c = 0; c < CC; ++c) {
            float v = ok ? xp[(size_t)c * HWSZ] : 0.0f;
            XtL[p*XT_STR + c] = (_Float16)v;
        }
    }
    __syncthreads();

    // ---------- phase 1: GEMM [256 pix x 64 K] x [64 K x 128 N] via WMMA f16 ----------
    {
        const int wid = tid >> 5, lane = tid & 31;
        const int hh  = lane >> 4, ln = lane & 15;

        #pragma unroll
        for (int mi = 0; mi < 2; ++mi) {
            const int mg = wid*2 + mi;            // M-tile 0..15
            const int m  = mg*16 + ln;            // pixel row this lane owns
            // A 16x32 per-lane layout: halves 0..7 = K[kb*32+hh*8 ..), halves 8..15 = +16
            AB16 A0, A1;
            const _Float16* a = XtL + m*XT_STR + hh*8;
            A0.u[0] = *(const uint4*)(a);        A0.u[1] = *(const uint4*)(a + 16);
            A1.u[0] = *(const uint4*)(a + 32);   A1.u[1] = *(const uint4*)(a + 48);

            for (int ng = 0; ng < 8; ++ng) {
                const int nrow = ng*16 + ln;      // N index this lane owns
                // B 32x16 per-lane layout: halves 0..7 = K[kb*32+hh*8 ..), halves 8..15 = +16
                AB16 B0, B1;
                const _Float16* bp = WtL + nrow*WT_STR + hh*8;
                B0.u[0] = *(const uint4*)(bp);       B0.u[1] = *(const uint4*)(bp + 16);
                B1.u[0] = *(const uint4*)(bp + 32);  B1.u[1] = *(const uint4*)(bp + 48);

                v8f acc = {0.f,0.f,0.f,0.f,0.f,0.f,0.f,0.f};
                acc = __builtin_amdgcn_wmma_f32_16x16x32_f16(false, A0.v, false, B0.v,
                                                             (short)0, acc, false, false);
                acc = __builtin_amdgcn_wmma_f32_16x16x32_f16(false, A1.v, false, B1.v,
                                                             (short)0, acc, false, false);

                const float bsv = biasL[nrow];
                #pragma unroll
                for (int r = 0; r < 8; ++r) {     // D: lane(n=ln,hh) vgpr r -> (M = r+8*hh, N = nrow)
                    const int p = mg*16 + r + 8*hh;
                    const _Float16 bv = (_Float16)(acc[r] + bsv);
                    if (nrow < MID)        X1L[p*X1_STR + nrow]           = bv;
                    else if (nrow < 2*MID) X2L[p*X1_STR + (nrow - MID)]   = bv;
                    else                   X3L[p*X3_STR + (nrow - 2*MID)] = bv;
                }
            }
        }
    }
    __syncthreads();

    // ---------- phase 2: 3x3 shifted sigmoid-attention ----------
    if (tid < TILE*TILE) {
        const int oy = tid / TILE, ox = tid % TILE;
        const int gh = ty*TILE + oy, gw = tx*TILE + ox;
        if (gh < HH && gw < WW) {
            const int pc = (oy + 1)*HALO + (ox + 1);
            float x2v[MID];
            #pragma unroll
            for (int c = 0; c < MID; ++c) x2v[c] = (float)X2L[pc*X1_STR + c];
            float acc[CC];
            #pragma unroll
            for (int c = 0; c < CC; ++c) acc[c] = 0.0f;

            for (int di = 0; di < 3; ++di) {
                for (int dj = 0; dj < 3; ++dj) {
                    const int nh = gh + di - 1, nw = gw + dj - 1;
                    if (nh < 0 || nh >= HH || nw < 0 || nw >= WW) continue; // x1s,x3s == 0
                    const int pn = (oy + di)*HALO + (ox + dj);
                    float attn = 0.0f;
                    #pragma unroll
                    for (int c = 0; c < MID; ++c) {
                        const float z = x2v[c] * (float)X1L[pn*X1_STR + c];
                        attn += 1.0f / (1.0f + __expf(-z));
                    }
                    #pragma unroll
                    for (int c = 0; c < CC; ++c)
                        acc[c] += attn * (float)X3L[pn*X3_STR + c];
                }
            }
            const float g = gamma[0];
            const size_t base = (size_t)bb*CC*HWSZ + (size_t)gh*WW + gw;
            #pragma unroll 4
            for (int c = 0; c < CC; ++c)
                out[base + (size_t)c*HWSZ] = x[base + (size_t)c*HWSZ] + g*acc[c];
        }
    }
}

extern "C" void kernel_launch(void* const* d_in, const int* in_sizes, int n_in,
                              void* d_out, int out_size, void* d_ws, size_t ws_size,
                              hipStream_t stream) {
    const float* x  = (const float*)d_in[0];
    const float* w1 = (const float*)d_in[1];
    const float* b1 = (const float*)d_in[2];
    const float* w2 = (const float*)d_in[3];
    const float* b2 = (const float*)d_in[4];
    const float* w3 = (const float*)d_in[5];
    const float* b3 = (const float*)d_in[6];
    const float* gm = (const float*)d_in[7];
    float* out = (float*)d_out;

    dim3 grid((WW + TILE - 1)/TILE, (HH + TILE - 1)/TILE, 4);
    dim3 block(256);
    fused_local_attn_gfx1250<<<grid, block, SMEM_BYTES, stream>>>(
        x, w1, b1, w2, b2, w3, b3, gm, out);
}
